// MultiHeadEncDecAttention_89472758710759
// MI455X (gfx1250) — compile-verified, matrix operand order checked by reading, not run on hardware
//
#include <hip/hip_runtime.h>
#include <hip/hip_bf16.h>

// ---------------------------------------------------------------------------
// CDNA5 (gfx1250) fused multi-head enc-dec attention, bf16 WMMA pipeline.
// wave32; all matrix math through v_wmma_f32_16x16x32_bf16.
// Operands pre-converted to bf16 once (packed hw convert when available).
// ---------------------------------------------------------------------------

typedef __attribute__((ext_vector_type(16))) __bf16 v16bf;
typedef __attribute__((ext_vector_type(8)))  float  v8f;

__device__ __forceinline__ unsigned short f32_to_bf16(float f) {
  union { float f; unsigned u; } c; c.f = f;
  unsigned r = c.u + 0x7FFFu + ((c.u >> 16) & 1u);   // round-to-nearest-even
  return (unsigned short)(r >> 16);
}

// pack two floats into two bf16 (one dword) — hw packed convert if present
__device__ __forceinline__ unsigned pack2_bf16(float lo, float hi) {
#if __has_builtin(__builtin_amdgcn_cvt_pk_bf16_f32)
  typedef __attribute__((ext_vector_type(2))) __bf16 v2bf;
  union { v2bf v; unsigned u; } c;
  c.v = __builtin_amdgcn_cvt_pk_bf16_f32(lo, hi);
  return c.u;
#else
  return (unsigned)f32_to_bf16(lo) | ((unsigned)f32_to_bf16(hi) << 16);
#endif
}

struct ABFrag { union { v16bf v; unsigned short h[16]; uint4 q[2]; }; };

// 16x32 bf16 A (or B^T) fragment from row-major [row][K] bf16 array.
// ISA layout: lane = row%16; halfwords 0..7 -> K = half*8+j ; 8..15 -> 16+half*8+j
__device__ __forceinline__ v16bf frag_bf16(const unsigned short* base, int stride,
                                           int k0, int lane) {
  const unsigned short* p = base + (lane & 15) * stride + k0 + (lane >> 4) * 8;
  ABFrag f;
  f.q[0] = *reinterpret_cast<const uint4*>(p);
  f.q[1] = *reinterpret_cast<const uint4*>(p + 16);
  return f.v;
}

__device__ __forceinline__ v8f wmma_bf16(v16bf a, v16bf b, v8f c) {
  return __builtin_amdgcn_wmma_f32_16x16x32_bf16(false, a, false, b,
                                                 (short)0, c, false, false);
}

// ---------------------------------------------------------------------------
// Kernel 0a: bulk fp32 -> bf16 (vectorized: float4 in, ushort4 out)
// ---------------------------------------------------------------------------
__global__ void k_cvt_bf16(const float* __restrict__ src,
                           unsigned short* __restrict__ dst, int nvec4) {
  int i = blockIdx.x * blockDim.x + threadIdx.x;
  if (i < nvec4) {
    float4 f = reinterpret_cast<const float4*>(src)[i];
    uint2 o;
    o.x = pack2_bf16(f.x, f.y);
    o.y = pack2_bf16(f.z, f.w);
    reinterpret_cast<uint2*>(dst)[i] = o;
  }
}

// ---------------------------------------------------------------------------
// Kernel 0b: z fp32 -> z_bf16 [g][t][d] and z_t bf16 [g][d][t]  (g=(h*8+v)*4+b)
// ---------------------------------------------------------------------------
__global__ void k_convert_z(const float* __restrict__ z,
                            unsigned short* __restrict__ zbf,
                            unsigned short* __restrict__ zt) {
  int g = blockIdx.x;
  const float*    src = z   + (size_t)g * (512 * 32);
  unsigned short* db  = zbf + (size_t)g * (512 * 32);
  unsigned short* dt  = zt  + (size_t)g * (512 * 32);
  // each thread: one (t, d-pair) -> 1 packed store + 2 transposed stores
  for (int i = threadIdx.x; i < 512 * 16; i += blockDim.x) {
    int t = i >> 4, dp = (i & 15) * 2;
    float lo = src[t * 32 + dp], hi = src[t * 32 + dp + 1];
    unsigned pk = pack2_bf16(lo, hi);
    reinterpret_cast<unsigned*>(db)[i] = pk;
    dt[(dp)     * 512 + t] = (unsigned short)(pk & 0xFFFF);
    dt[(dp + 1) * 512 + t] = (unsigned short)(pk >> 16);
  }
}

// ---------------------------------------------------------------------------
// Kernel 1: q = x @ w_q^T + b, pre-scaled by 1/sqrt(D_Q),
//           stored bf16 in [h][v][b][s][32] layout. 8 WMMAs per wave.
// ---------------------------------------------------------------------------
__global__ void __launch_bounds__(128)
k_qproj(const unsigned short* __restrict__ xbf,
        const unsigned short* __restrict__ wqbf,
        const float* __restrict__ bq, unsigned short* __restrict__ qbf) {
  int lane = threadIdx.x & 31;
  int wave = threadIdx.x >> 5;
  int mt = blockIdx.x >> 2;                 // 1024 m-tiles (16384 rows)
  int nt = (blockIdx.x & 3) * 4 + wave;     // 16 n-tiles (256 cols)
  const unsigned short* abase = xbf  + (size_t)mt * 16 * 256;
  const unsigned short* bbase = wqbf + (size_t)nt * 16 * 256;  // B^T row = wq row
  v8f acc = {};
#pragma unroll
  for (int kc = 0; kc < 8; ++kc) {
    v16bf a = frag_bf16(abase, 256, kc * 32, lane);
    v16bf b = frag_bf16(bbase, 256, kc * 32, lane);
    acc = wmma_bf16(a, b, acc);
  }
  int n = lane & 15, hf = lane >> 4;
  int e = nt * 16 + n;                      // output feature = h*32 + d
  const float scale = 0.17677669529663687f; // 1/sqrt(32), folded into q
  float bias = bq[e];
  int hh = e >> 5, dd = e & 31;
#pragma unroll
  for (int i = 0; i < 8; ++i) {
    int r  = mt * 16 + hf * 8 + i;          // r = ((b*512+s)*8+v)
    int vv = r & 7, ss = (r >> 3) & 511, bb = r >> 12;
    int g  = (hh * 8 + vv) * 4 + bb;
    qbf[((size_t)g * 512 + ss) * 32 + dd] = f32_to_bf16((acc[i] + bias) * scale);
  }
}

// ---------------------------------------------------------------------------
// Kernel 2: per (g, 16-row s-tile) attention. One wave per block.
//   S = (Q/sqrt(d)) Z^T   (32 WMMAs, K = d = 32, fully unrolled)
//   softmax over t = 512 in LDS (wave32 cross-half shuffle reduce)
//   O = P Z               (32 WMMAs, fully unrolled, 2 d-tiles)
// ---------------------------------------------------------------------------
__global__ void __launch_bounds__(32)
k_attn(const unsigned short* __restrict__ qbf,
       const unsigned short* __restrict__ zbf,
       const unsigned short* __restrict__ zt,
       unsigned short* __restrict__ abf) {
  __shared__ __align__(16) float          Ssc[16 * 512];   // 32 KB
  __shared__ __align__(16) unsigned short Pbf[16 * 512];   // 16 KB
  int lane  = threadIdx.x & 31;
  int g     = blockIdx.x >> 5;
  int stile = blockIdx.x & 31;
  const unsigned short* qb  = qbf + ((size_t)g * 512 + stile * 16) * 32;
  const unsigned short* zb  = zbf + (size_t)g * 512 * 32;
  const unsigned short* ztb = zt  + (size_t)g * 512 * 32;
  int n = lane & 15, hf = lane >> 4;

  v16bf aq = frag_bf16(qb, 32, 0, lane);
  __builtin_prefetch(zb, 0, 1);                            // global_prefetch_b8
#pragma unroll
  for (int tc = 0; tc < 32; ++tc) {
    v16bf bz = frag_bf16(zb + tc * 16 * 32, 32, 0, lane);  // B^T row = z row (t)
    v8f c = {};
    c = wmma_bf16(aq, bz, c);
#pragma unroll
    for (int i = 0; i < 8; ++i)
      Ssc[(hf * 8 + i) * 512 + tc * 16 + n] = c[i];
  }
  __syncthreads();

  // softmax: lane pair (L, L^16) shares row (lane&15), 256 cols each half
  {
    float* row = Ssc + (lane & 15) * 512 + hf * 256;
    float m = -3.0e38f;
    for (int j = 0; j < 256; ++j) m = fmaxf(m, row[j]);
    m = fmaxf(m, __shfl_xor(m, 16, 32));
    float sum = 0.f;
    for (int j = 0; j < 256; ++j) { float e = __expf(row[j] - m); row[j] = e; sum += e; }
    sum += __shfl_xor(sum, 16, 32);
    float inv = 1.0f / sum;
    unsigned* prow = reinterpret_cast<unsigned*>(Pbf + (lane & 15) * 512 + hf * 256);
    for (int j = 0; j < 128; ++j)
      prow[j] = pack2_bf16(row[2 * j] * inv, row[2 * j + 1] * inv);
  }
  __syncthreads();

  v8f o0 = {}, o1 = {};
#pragma unroll
  for (int tc = 0; tc < 16; ++tc) {
    v16bf ap = frag_bf16(Pbf, 512, tc * 32, lane);                  // A: 16x32 of P
    v16bf b0 = frag_bf16(ztb,            512, tc * 32, lane);       // B^T rows = z^T
    v16bf b1 = frag_bf16(ztb + 16 * 512, 512, tc * 32, lane);
    o0 = wmma_bf16(ap, b0, o0);
    o1 = wmma_bf16(ap, b1, o1);
  }
  // out[h,v,b,s,d] -> abf[((b*512+s)*8+v)*256 + h*32 + d]
  int hh = g >> 5, vv = (g >> 2) & 7, bb = g & 3;
#pragma unroll
  for (int i = 0; i < 8; ++i) {
    int s = stile * 16 + hf * 8 + i;
    size_t r = ((size_t)(bb * 512 + s) * 8 + vv);
    abf[r * 256 + hh * 32 + n]      = f32_to_bf16(o0[i]);
    abf[r * 256 + hh * 32 + 16 + n] = f32_to_bf16(o1[i]);
  }
}

// ---------------------------------------------------------------------------
// Kernel 3: y = attn_out @ fc_w^T + fc_b + x ; LayerNorm over 256.
// 16 waves/block: wave w owns n-tile w (GEMM, 8 WMMAs) then row w (LN).
// ---------------------------------------------------------------------------
__global__ void __launch_bounds__(512)
k_fc_ln(const unsigned short* __restrict__ abf,
        const unsigned short* __restrict__ fwbf, const float* __restrict__ fcb,
        const float* __restrict__ x,   const float* __restrict__ gamma,
        const float* __restrict__ beta, float* __restrict__ out) {
  __shared__ __align__(16) float Y[16 * 256];              // 16 KB
  int lane = threadIdx.x & 31;
  int wave = threadIdx.x >> 5;                             // n-tile / LN row
  int mt = blockIdx.x;
  const unsigned short* abase = abf  + (size_t)mt * 16 * 256;
  const unsigned short* bbase = fwbf + (size_t)wave * 16 * 256;  // fc_w[dcol][e]
  v8f acc = {};
#pragma unroll
  for (int kc = 0; kc < 8; ++kc) {
    v16bf a = frag_bf16(abase, 256, kc * 32, lane);
    v16bf b = frag_bf16(bbase, 256, kc * 32, lane);
    acc = wmma_bf16(a, b, acc);
  }
  int n = lane & 15, hf = lane >> 4;
  int col = wave * 16 + n;
  float bias = fcb[col];
#pragma unroll
  for (int i = 0; i < 8; ++i) {
    int rloc = hf * 8 + i;
    int r = mt * 16 + rloc;
    Y[rloc * 256 + col] = acc[i] + bias + x[(size_t)r * 256 + col];
  }
  __syncthreads();

  float s = 0.f, s2 = 0.f, vals[8];
#pragma unroll
  for (int k = 0; k < 8; ++k) {
    float v = Y[wave * 256 + lane + k * 32];
    vals[k] = v; s += v; s2 += v * v;
  }
#pragma unroll
  for (int off = 16; off >= 1; off >>= 1) {
    s  += __shfl_xor(s,  off, 32);
    s2 += __shfl_xor(s2, off, 32);
  }
  float mu   = s  * (1.0f / 256.0f);
  float var  = s2 * (1.0f / 256.0f) - mu * mu;
  float rstd = rsqrtf(var + 1e-5f);
  int r = mt * 16 + wave;
#pragma unroll
  for (int k = 0; k < 8; ++k) {
    int c = lane + k * 32;
    out[(size_t)r * 256 + c] = (vals[k] - mu) * rstd * gamma[c] + beta[c];
  }
}

// ---------------------------------------------------------------------------
extern "C" void kernel_launch(void* const* d_in, const int* in_sizes, int n_in,
                              void* d_out, int out_size, void* d_ws, size_t ws_size,
                              hipStream_t stream) {
  const float* x  = (const float*)d_in[0];
  const float* z  = (const float*)d_in[1];
  const float* wq = (const float*)d_in[2];
  const float* bq = (const float*)d_in[3];
  const float* fw = (const float*)d_in[4];
  const float* fb = (const float*)d_in[5];
  const float* gm = (const float*)d_in[6];
  const float* bt = (const float*)d_in[7];

  // workspace layout (bf16 elements)
  unsigned short* qbf  = (unsigned short*)d_ws;             // 4M  q  [h][v][b][s][32]
  unsigned short* zbf  = qbf  + (size_t)4 * 1024 * 1024;    // 4M  z  [g][t][32]
  unsigned short* zt   = zbf  + (size_t)4 * 1024 * 1024;    // 4M  z^T[g][32][t]
  unsigned short* abf  = zt   + (size_t)4 * 1024 * 1024;    // 4M  attn out [r][256]
  unsigned short* xbf  = abf  + (size_t)4 * 1024 * 1024;    // 4M  x  [r][256]
  unsigned short* wqbf = xbf  + (size_t)4 * 1024 * 1024;    // 64K w_q [e][256]
  unsigned short* fwbf = wqbf + (size_t)64 * 1024;          // 64K fc_w[d][256]

  k_cvt_bf16<<<4096, 256, 0, stream>>>(x,  xbf,  4 * 1024 * 1024 / 4);
  k_cvt_bf16<<<64,   256, 0, stream>>>(wq, wqbf, 65536 / 4);
  k_cvt_bf16<<<64,   256, 0, stream>>>(fw, fwbf, 65536 / 4);
  k_convert_z<<<256, 256, 0, stream>>>(z, zbf, zt);
  k_qproj   <<<4096, 128, 0, stream>>>(xbf, wqbf, bq, qbf);
  k_attn    <<<8192, 32, 0, stream>>>(qbf, zbf, zt, abf);
  k_fc_ln   <<<1024, 512, 0, stream>>>(abf, fwbf, fb, x, gm, bt, (float*)d_out);
}
